// SelfAttention_64089501991342
// MI455X (gfx1250) — matrix operand order repaired and checked by reading, not compiled, not run on hardware
//
#include <hip/hip_runtime.h>
#include <hip/hip_bf16.h>

#define BATCH 8
#define CH    512
#define NN    2048

typedef __attribute__((ext_vector_type(16))) __bf16 v16bf;
typedef __attribute__((ext_vector_type(8)))  __bf16 v8bf;
typedef __attribute__((ext_vector_type(8)))  float  v8f;

static __device__ __forceinline__ __bf16 f2bf(float f) {
  unsigned u = __float_as_uint(f);
  u = (u + 0x7FFFu + ((u >> 16) & 1u)) >> 16;   // round-to-nearest-even
  unsigned short s = (unsigned short)u;
  __bf16 b;
  __builtin_memcpy(&b, &s, sizeof(b));
  return b;
}

static __device__ __forceinline__ v16bf combine8(v8bf lo, v8bf hi) {
  v16bf a;
#pragma unroll
  for (int e = 0; e < 8; ++e) { a[e] = lo[e]; a[e + 8] = hi[e]; }
  return a;
}

// A-matrix 16x32 bf16 fragment: lane-half h -> K chunks [h*8..h*8+7], [h*8+16..h*8+23]
static __device__ __forceinline__ v16bf load_a_frag(const __bf16* row, int h) {
  return combine8(*(const v8bf*)(row + h * 8), *(const v8bf*)(row + h * 8 + 16));
}

// B-matrix 32x16 bf16 fragment: lane-half h -> K chunks [h*8..h*8+7], [h*8+16..h*8+23] for its column.
static __device__ __forceinline__ v16bf load_b_frag(const __bf16* row, int h) {
  return combine8(*(const v8bf*)(row + h * 8), *(const v8bf*)(row + h * 8 + 16));
}

static __device__ __forceinline__ v8f wmma_bf16(v16bf a, v16bf b, v8f c) {
  return __builtin_amdgcn_wmma_f32_16x16x32_bf16(false, a, false, b, (short)0, c,
                                                 false, false);
}

// ---------- pack: f32 -> bf16, elementwise (weights) ----------
__global__ __launch_bounds__(256) void pack_f32_bf16(const float* __restrict__ in,
                                                     __bf16* __restrict__ out, int n) {
  int i = (blockIdx.x * blockDim.x + threadIdx.x) * 4;
  if (i + 3 < n) {
    float4 v = *(const float4*)(in + i);
    out[i + 0] = f2bf(v.x);
    out[i + 1] = f2bf(v.y);
    out[i + 2] = f2bf(v.z);
    out[i + 3] = f2bf(v.w);
  }
}

// ---------- transpose+pack: x [B][C][N] f32 -> xT [B][N][C] bf16 ----------
__global__ __launch_bounds__(256) void transpose_pack_kernel(const float* __restrict__ x,
                                                             __bf16* __restrict__ xT) {
  __shared__ float tile[32][33];
  const int b = blockIdx.z;
  const int n0 = blockIdx.x * 32;
  const int c0 = blockIdx.y * 32;
  const int tx = threadIdx.x, ty = threadIdx.y;   // 32 x 8
#pragma unroll
  for (int k = 0; k < 4; ++k) {
    int c = c0 + ty + k * 8;
    tile[ty + k * 8][tx] = x[((size_t)b * CH + c) * NN + n0 + tx];
  }
  __syncthreads();
#pragma unroll
  for (int k = 0; k < 4; ++k) {
    int n = n0 + ty + k * 8;
    xT[((size_t)b * NN + n) * CH + c0 + tx] = f2bf(tile[tx][ty + k * 8]);
  }
}

// ---------- Q/K projection, 2x4 register-blocked:
// D[o][n] = sum_c W[o][c]*X[c][n] + b[o], stored transposed into outT [B][N][C] ----------
__global__ __launch_bounds__(256) void proj_nt_kernel(const __bf16* __restrict__ wbf,
                                                      const float* __restrict__ bias,
                                                      const __bf16* __restrict__ xT,
                                                      __bf16* __restrict__ outT) {
  const int lane = threadIdx.x & 31;
  const int wv = threadIdx.x >> 5;
  const int h = lane >> 4, l15 = lane & 15;
  const int gw = blockIdx.x * 8 + wv;       // 0..511 per batch
  const int b = blockIdx.y;
  const int o0 = (gw & 15) * 32;            // 2 o-tiles  (CH/32 = 16 groups)
  const int n0 = (gw >> 4) * 64;            // 4 n-tiles  (NN/64 = 32 groups)

  const __bf16* arow[2];
  const __bf16* brow[4];
#pragma unroll
  for (int i = 0; i < 2; ++i) arow[i] = wbf + (size_t)(o0 + i * 16 + l15) * CH;
#pragma unroll
  for (int j = 0; j < 4; ++j) brow[j] = xT + ((size_t)b * NN + n0 + j * 16 + l15) * CH;

  v8f acc[2][4] = {};
#pragma unroll 2
  for (int ks = 0; ks < CH / 32; ++ks) {
    v16bf af[2], bf_[4];
#pragma unroll
    for (int i = 0; i < 2; ++i) af[i] = load_a_frag(arow[i] + ks * 32, h);
#pragma unroll
    for (int j = 0; j < 4; ++j) bf_[j] = load_b_frag(brow[j] + ks * 32, h);
#pragma unroll
    for (int i = 0; i < 2; ++i)
#pragma unroll
      for (int j = 0; j < 4; ++j) acc[i][j] = wmma_bf16(af[i], bf_[j], acc[i][j]);
  }

#pragma unroll
  for (int i = 0; i < 2; ++i) {
    const float* bp = bias + o0 + i * 16 + 8 * h;   // o contiguous over r
#pragma unroll
    for (int j = 0; j < 4; ++j) {
      __bf16* op = outT + ((size_t)b * NN + n0 + j * 16 + l15) * CH + o0 + i * 16 + 8 * h;
      v8bf res;
#pragma unroll
      for (int r = 0; r < 8; ++r) res[r] = f2bf(acc[i][j][r] + bp[r]);
      *(v8bf*)op = res;
    }
  }
}

// ---------- V projection (swapped form), 2x4 register-blocked:
// D[n][o] = sum_c Xt[n][c]*W[o][c] + b[o], stored into outV [B][C][N] ----------
__global__ __launch_bounds__(256) void proj_cn_kernel(const __bf16* __restrict__ wbf,
                                                      const float* __restrict__ bias,
                                                      const __bf16* __restrict__ xT,
                                                      __bf16* __restrict__ outV) {
  const int lane = threadIdx.x & 31;
  const int wv = threadIdx.x >> 5;
  const int h = lane >> 4, l15 = lane & 15;
  const int gw = blockIdx.x * 8 + wv;       // 0..511 per batch
  const int b = blockIdx.y;
  const int n0 = (gw & 63) * 32;            // 2 n-tiles  (NN/32 = 64 groups)
  const int o0 = (gw >> 6) * 64;            // 4 o-tiles  (CH/64 =  8 groups)

  const __bf16* arow[2];
  const __bf16* brow[4];
#pragma unroll
  for (int i = 0; i < 2; ++i) arow[i] = xT + ((size_t)b * NN + n0 + i * 16 + l15) * CH;
#pragma unroll
  for (int j = 0; j < 4; ++j) brow[j] = wbf + (size_t)(o0 + j * 16 + l15) * CH;

  v8f acc[2][4] = {};
#pragma unroll 2
  for (int ks = 0; ks < CH / 32; ++ks) {
    v16bf af[2], bf_[4];
#pragma unroll
    for (int i = 0; i < 2; ++i) af[i] = load_a_frag(arow[i] + ks * 32, h);
#pragma unroll
    for (int j = 0; j < 4; ++j) bf_[j] = load_b_frag(brow[j] + ks * 32, h);
#pragma unroll
    for (int i = 0; i < 2; ++i)
#pragma unroll
      for (int j = 0; j < 4; ++j) acc[i][j] = wmma_bf16(af[i], bf_[j], acc[i][j]);
  }

#pragma unroll
  for (int j = 0; j < 4; ++j) {
    const int o = o0 + j * 16 + l15;        // fixed per lane
    const float bb = bias[o];
#pragma unroll
    for (int i = 0; i < 2; ++i) {
      __bf16* op = outV + ((size_t)b * CH + o) * NN + n0 + i * 16 + 8 * h;  // n contig
      v8bf res;
#pragma unroll
      for (int r = 0; r < 8; ++r) res[r] = f2bf(acc[i][j][r] + bb);
      *(v8bf*)op = res;
    }
  }
}

// ---------- fused attention: one (b, i-tile) per workgroup, scores strip in LDS ----------
__global__ __launch_bounds__(256) void attn_kernel(const __bf16* __restrict__ qT,
                                                   const __bf16* __restrict__ kT,
                                                   const __bf16* __restrict__ vB,
                                                   const float* __restrict__ x,
                                                   float* __restrict__ out) {
  extern __shared__ char smem[];
  float* S = (float*)smem;                               // 16 x NN f32  (128 KB)
  __bf16* P = (__bf16*)(smem + (size_t)16 * NN * 4);     // 16 x NN bf16 ( 64 KB)
  const int lane = threadIdx.x & 31;
  const int w = threadIdx.x >> 5;
  const int h = lane >> 4, l15 = lane & 15;
  const int b = blockIdx.y;
  const int i0 = blockIdx.x * 16;

  // Phase 1: S[i][j] = sum_c q[c][i] k[c][j]; A (q rows) shared across 4 j-tiles
  const __bf16* qrow = qT + ((size_t)b * NN + i0 + l15) * CH;
  for (int g = w; g < 32; g += 8) {         // 128 j-tiles in groups of 4
    const int jt0 = g * 4;
    const __bf16* krow[4];
#pragma unroll
    for (int jj = 0; jj < 4; ++jj)
      krow[jj] = kT + ((size_t)b * NN + (jt0 + jj) * 16 + l15) * CH;
    v8f acc[4] = {};
#pragma unroll 2
    for (int ks = 0; ks < CH / 32; ++ks) {
      v16bf a = load_a_frag(qrow + ks * 32, h);
#pragma unroll
      for (int jj = 0; jj < 4; ++jj) {
        v16bf bm = load_b_frag(krow[jj] + ks * 32, h);
        acc[jj] = wmma_bf16(a, bm, acc[jj]);
      }
    }
#pragma unroll
    for (int jj = 0; jj < 4; ++jj)
#pragma unroll
      for (int r = 0; r < 8; ++r)
        S[(size_t)(r + 8 * h) * NN + (jt0 + jj) * 16 + l15] = acc[jj][r];
  }
  __syncthreads();

  // Phase 2: exact softmax over j, per row; wave w owns rows 2w and 2w+1
#pragma unroll
  for (int rr = 0; rr < 2; ++rr) {
    const int row = w * 2 + rr;
    float* sr = S + (size_t)row * NN;
    float m = -1e30f;
    for (int j = lane; j < NN; j += 32) m = fmaxf(m, sr[j]);
#pragma unroll
    for (int off = 16; off > 0; off >>= 1) m = fmaxf(m, __shfl_xor(m, off, 32));
    float sum = 0.f;
    for (int j = lane; j < NN; j += 32) {
      float e = __expf(sr[j] - m);
      sr[j] = e;
      sum += e;
    }
#pragma unroll
    for (int off = 16; off > 0; off >>= 1) sum += __shfl_xor(sum, off, 32);
    const float rinv = 1.0f / sum;
    __bf16* pr = P + (size_t)row * NN;
    for (int j = lane; j < NN; j += 32) pr[j] = f2bf(sr[j] * rinv);
  }
  __syncthreads();

  // Phase 3: out[c][i] = sum_j P[i][j]*v[c][j] + x[c][i]; 4 c-tiles per wave
  const __bf16* prow = P + (size_t)l15 * NN;    // A rows: i (LDS)
  {
    const int ct0 = w * 4;                      // 32 c-tiles / 8 waves
    const __bf16* vrow[4];
#pragma unroll
    for (int cc = 0; cc < 4; ++cc)
      vrow[cc] = vB + ((size_t)b * CH + (ct0 + cc) * 16 + l15) * NN;
    v8f acc[4] = {};
#pragma unroll 2
    for (int ks = 0; ks < NN / 32; ++ks) {
      v16bf a = load_a_frag(prow + ks * 32, h); // ds_load_b128 x2
#pragma unroll
      for (int cc = 0; cc < 4; ++cc) {
        v16bf bm = load_b_frag(vrow[cc] + ks * 32, h);
        acc[cc] = wmma_bf16(a, bm, acc[cc]);
      }
    }
#pragma unroll
    for (int cc = 0; cc < 4; ++cc) {
      const int c = (ct0 + cc) * 16 + l15;      // fixed per lane
      const size_t base = ((size_t)b * CH + c) * NN + i0 + 8 * h;  // i contiguous
      const float* xr = x + base;
      float* orow = out + base;
#pragma unroll
      for (int r = 0; r < 8; ++r) orow[r] = acc[cc][r] + xr[r];
    }
  }
}

extern "C" void kernel_launch(void* const* d_in, const int* in_sizes, int n_in,
                              void* d_out, int out_size, void* d_ws, size_t ws_size,
                              hipStream_t stream) {
  (void)in_sizes; (void)n_in; (void)out_size; (void)ws_size;
  const float* x  = (const float*)d_in[0];
  const float* wq = (const float*)d_in[1];
  const float* bq = (const float*)d_in[2];
  const float* wk = (const float*)d_in[3];
  const float* bk = (const float*)d_in[4];
  const float* wv = (const float*)d_in[5];
  const float* bv = (const float*)d_in[6];
  float* out = (float*)d_out;

  char* ws = (char*)d_ws;
  const size_t big = (size_t)BATCH * NN * CH * sizeof(__bf16);   // 16 MiB
  const size_t wsz = (size_t)CH * CH * sizeof(__bf16);           // 512 KiB
  __bf16* xT  = (__bf16*)(ws);
  __bf16* qT  = (__bf16*)(ws + big);
  __bf16* kT  = (__bf16*)(ws + 2 * big);
  __bf16* vB  = (__bf16*)(ws + 3 * big);
  __bf16* wqb = (__bf16*)(ws + 4 * big);
  __bf16* wkb = (__bf16*)(ws + 4 * big + wsz);
  __bf16* wvb = (__bf16*)(ws + 4 * big + 2 * wsz);

  const int wn = CH * CH;                                  // 262144
  pack_f32_bf16<<<wn / 1024, 256, 0, stream>>>(wq, wqb, wn);
  pack_f32_bf16<<<wn / 1024, 256, 0, stream>>>(wk, wkb, wn);
  pack_f32_bf16<<<wn / 1024, 256, 0, stream>>>(wv, wvb, wn);

  transpose_pack_kernel<<<dim3(NN / 32, CH / 32, BATCH), dim3(32, 8), 0, stream>>>(x, xT);

  // 512 wave-tiles per batch, 8 waves per block -> 64 blocks per batch
  dim3 pg(64, BATCH);
  proj_nt_kernel<<<pg, 256, 0, stream>>>(wqb, bq, xT, qT);
  proj_nt_kernel<<<pg, 256, 0, stream>>>(wkb, bk, xT, kT);
  proj_cn_kernel<<<pg, 256, 0, stream>>>(wvb, bv, xT, vB);

  const size_t lds = (size_t)16 * NN * 4 + (size_t)16 * NN * 2;  // 192 KB (< 320 KB/WGP)
  attn_kernel<<<dim3(NN / 16, BATCH), 256, lds, stream>>>(qT, kT, vB, x, out);
}
